// HierarchicalMemoryDNC_850403525345
// MI455X (gfx1250) — matrix operand order repaired and checked by reading, not compiled, not run on hardware
//
#include <hip/hip_runtime.h>
#include <hip/hip_bf16.h>
#include <cstddef>

// ---- problem constants ------------------------------------------------------
#define BB   2
#define TSEQ 2048
#define DD   512
#define HH   8
#define LL   4
#define HD   64
#define MTOT (BB * TSEQ)      // 4096 token rows
#define VOC  32000

typedef __bf16 bf16;
typedef __attribute__((ext_vector_type(16))) __bf16 bf16x16;
typedef __attribute__((ext_vector_type(8)))  float  floatx8;
typedef __attribute__((ext_vector_type(4))) unsigned uint4v;
typedef __attribute__((ext_vector_type(8))) unsigned uint8v;

// ---- WMMA wrapper -----------------------------------------------------------
static __device__ inline floatx8 wmma_bf16(bf16x16 a, bf16x16 b, floatx8 c) {
  // v_wmma_f32_16x16x32_bf16  (D = A[16x32] * B[32x16] + C)
  return __builtin_amdgcn_wmma_f32_16x16x32_bf16(false, a, false, b,
                                                 (short)0, c, false, false);
}

// ---- fragment loaders (wave32; contiguous k order, element s holds k = 16*half + s) -----------------
// A fragment: 16x32 bf16, row-major source with leading dim lda.
// lane m = lane&15 owns row m; half = lane>>4 selects K sub-blocks:
// VGPR v pair covers K = 2*v + 16*half .. +1
static __device__ inline bf16x16 load_a_bf16(const bf16* __restrict__ A, int lda,
                                             int row0, int k0) {
  int lane = threadIdx.x & 31;
  int m = lane & 15, half = lane >> 4;
  const bf16* base = A + (size_t)(row0 + m) * lda + k0;
  bf16x16 a;
#pragma unroll
  for (int v = 0; v < 8; ++v) {
    int kb = 2 * v + 16 * half;
    a[2 * v]     = base[kb];
    a[2 * v + 1] = base[kb + 1];
  }
  return a;
}

// B fragment = W^T where W is [N,K] row-major fp32 (converted on the fly).
// lane n = lane&15 owns column n; element e covers K = e + 16*half.
static __device__ inline bf16x16 load_bT_f32(const float* __restrict__ W, int ldw,
                                             int col0, int k0) {
  int lane = threadIdx.x & 31;
  int n = lane & 15, half = lane >> 4;
  const float* base = W + (size_t)(col0 + n) * ldw + k0 + 16 * half;
  bf16x16 b;
#pragma unroll
  for (int e = 0; e < 16; ++e) b[e] = (bf16)base[e];
  return b;
}

// Same pattern, bf16 source (attention K^T).
static __device__ inline bf16x16 load_bT_bf16(const bf16* __restrict__ W, int ldw,
                                              int col0, int k0) {
  int lane = threadIdx.x & 31;
  int n = lane & 15, half = lane >> 4;
  const bf16* base = W + (size_t)(col0 + n) * ldw + k0 + 16 * half;
  bf16x16 b;
#pragma unroll
  for (int e = 0; e < 16; ++e) b[e] = base[e];
  return b;
}

// B fragment from row-major [K,N] bf16 (attention V): B[k][n] = Vm[k][n].
static __device__ inline bf16x16 load_b_rm_bf16(const bf16* __restrict__ Vm, int ldn,
                                                int k0, int col0) {
  int lane = threadIdx.x & 31;
  int n = lane & 15, half = lane >> 4;
  bf16x16 b;
#pragma unroll
  for (int e = 0; e < 16; ++e)
    b[e] = Vm[(size_t)(k0 + 16 * half + e) * ldn + col0 + n];
  return b;
}

// ---- embedding --------------------------------------------------------------
__global__ void embed_kernel(const int* __restrict__ ids, const float* __restrict__ tok,
                             const float* __restrict__ pos, float* __restrict__ xf,
                             bf16* __restrict__ xb) {
  int row = blockIdx.x;
  int t = row & (TSEQ - 1);
  int id = ids[row];
  for (int d = threadIdx.x; d < DD; d += blockDim.x) {
    float v = tok[(size_t)id * DD + d] + pos[(size_t)t * DD + d];
    xf[(size_t)row * DD + d] = v;
    xb[(size_t)row * DD + d] = (bf16)v;
  }
}

// ---- generic WMMA GEMM: C = A[M,K](bf16) @ W[N,K]^T(f32) + bias -------------
// The 16-row A tile (16*K bf16, fully contiguous) is DMAed into LDS once per
// block by the Tensor Data Mover; all 8 waves then read fragments from LDS.
// epilogue: optional exact GELU, optional residual add, f32 and/or bf16 out.
template <int K>
__global__ __launch_bounds__(256, 1)
void gemm_kernel(const bf16* __restrict__ A, const float* __restrict__ W,
                 const float* __restrict__ bias, const float* __restrict__ residual,
                 float* __restrict__ outF, bf16* __restrict__ outB,
                 int N, int act) {
  __shared__ bf16 Atile[16 * K];
  int wave = threadIdx.x >> 5, lane = threadIdx.x & 31;
  int n = lane & 15, half = lane >> 4;
  int col0 = (blockIdx.x * 8 + wave) * 16;
  int row0 = blockIdx.y * 16;

  // --- TDM stage: one tensor_load_to_lds per block (wave 0, scalar branch) ---
  int wid = __builtin_amdgcn_readfirstlane(threadIdx.x >> 5);
  if (wid == 0) {
    unsigned ldsa = (unsigned)(size_t)(void*)&Atile[0];       // LDS byte address
    unsigned long long ga =
        (unsigned long long)(size_t)(A + (size_t)row0 * K);   // global byte address
    unsigned ndw = (unsigned)((16 * K * 2) / 4);              // tile length in dwords
    uint4v g0;
    g0[0] = 1u;                                   // count=1, user mode, no gather
    g0[1] = ldsa;                                 // lds_addr
    g0[2] = (unsigned)ga;                         // global_addr[31:0]
    g0[3] = (unsigned)(ga >> 32) | (2u << 30);    // global_addr[56:32] | type=2
    uint8v g1;
    g1[0] = (2u << 16);                           // data_size=2 (4B), no multicast
    g1[1] = (ndw & 0xFFFFu) << 16;                // tensor_dim0[15:0] -> bits 63:48
    g1[2] = ndw >> 16;                            // tensor_dim0[31:16]
    g1[3] = (ndw & 0xFFFFu) << 16;                // tile_dim0 -> bits 127:112
    g1[4] = 0u; g1[5] = 0u; g1[6] = 0u; g1[7] = 0u;
    asm volatile("tensor_load_to_lds %0, %1" :: "s"(g0), "s"(g1) : "memory");
    asm volatile("s_wait_tensorcnt 0x0" ::: "memory");
  }
  __syncthreads();

  floatx8 acc = {0, 0, 0, 0, 0, 0, 0, 0};
  for (int k0 = 0; k0 < K; k0 += 32) {
    if (k0 + 64 < K)  // prefetch streamed W operand two steps ahead
      __builtin_prefetch(W + (size_t)(col0 + n) * K + k0 + 64, 0, 1);
    acc = wmma_bf16(load_a_bf16(Atile, K, 0, k0), load_bT_f32(W, K, col0, k0), acc);
  }
  float bv = bias ? bias[col0 + n] : 0.f;
#pragma unroll
  for (int v = 0; v < 8; ++v) {
    int m = v + 8 * half;
    float x = acc[v] + bv;
    if (act) x = 0.5f * x * (1.f + erff(x * 0.70710678f));  // exact gelu
    size_t idx = (size_t)(row0 + m) * N + col0 + n;
    if (residual) x += residual[idx];
    if (outF) outF[idx] = x;
    if (outB) outB[idx] = (bf16)x;
  }
}

// ---- flash attention: one wave = 16 queries of one (b,h) --------------------
__global__ __launch_bounds__(256, 1)
void attn_kernel(const bf16* __restrict__ qkvb, bf16* __restrict__ ob) {
  __shared__ bf16 Pl[8][16 * 32];  // per-wave P tile (16 queries x 32 keys)
  int wave = threadIdx.x >> 5, lane = threadIdx.x & 31;
  int n = lane & 15, half = lane >> 4;
  int qt = blockIdx.x * 8 + wave;           // 0..127
  int b = blockIdx.y >> 3, h = blockIdx.y & 7;
  const bf16* base = qkvb + (size_t)b * TSEQ * (3 * DD);
  const bf16* Qp = base + h * HD;
  const bf16* Kp = base + DD + h * HD;
  const bf16* Vp = base + 2 * DD + h * HD;
  int q0 = qt * 16;

  bf16x16 qf0 = load_a_bf16(Qp, 3 * DD, q0, 0);
  bf16x16 qf1 = load_a_bf16(Qp, 3 * DD, q0, 32);

  floatx8 o0 = {0, 0, 0, 0, 0, 0, 0, 0}, o1 = o0, o2 = o0, o3 = o0;
  float mm[8], ls[8];
#pragma unroll
  for (int v = 0; v < 8; ++v) { mm[v] = -3e38f; ls[v] = 0.f; }

  for (int kb = 0; kb < TSEQ; kb += 32) {
    floatx8 sA = {0, 0, 0, 0, 0, 0, 0, 0}, sB = sA;
    sA = wmma_bf16(qf0, load_bT_bf16(Kp, 3 * DD, kb, 0), sA);
    sA = wmma_bf16(qf1, load_bT_bf16(Kp, 3 * DD, kb, 32), sA);
    sB = wmma_bf16(qf0, load_bT_bf16(Kp, 3 * DD, kb + 16, 0), sB);
    sB = wmma_bf16(qf1, load_bT_bf16(Kp, 3 * DD, kb + 16, 32), sB);
#pragma unroll
    for (int v = 0; v < 8; ++v) {
      float x0 = sA[v] * 0.125f, x1 = sB[v] * 0.125f;  // 1/sqrt(64)
      float t = fmaxf(x0, x1);
      t = fmaxf(t, __shfl_xor(t, 1, 32));
      t = fmaxf(t, __shfl_xor(t, 2, 32));
      t = fmaxf(t, __shfl_xor(t, 4, 32));
      t = fmaxf(t, __shfl_xor(t, 8, 32));   // row max within 16-lane half
      float mn = fmaxf(mm[v], t);
      float sc = __expf(mm[v] - mn);
      mm[v] = mn;
      float p0 = __expf(x0 - mn), p1 = __expf(x1 - mn);
      float ps = p0 + p1;
      ps += __shfl_xor(ps, 1, 32);
      ps += __shfl_xor(ps, 2, 32);
      ps += __shfl_xor(ps, 4, 32);
      ps += __shfl_xor(ps, 8, 32);
      ls[v] = ls[v] * sc + ps;
      o0[v] = o0[v] * sc; o1[v] = o1[v] * sc; o2[v] = o2[v] * sc; o3[v] = o3[v] * sc;
      int m = v + 8 * half;
      Pl[wave][m * 32 + n]      = (bf16)p0;
      Pl[wave][m * 32 + 16 + n] = (bf16)p1;
    }
    asm volatile("s_wait_dscnt 0" ::: "memory");  // P stores visible (DS in order per wave)
    bf16x16 pa = load_a_bf16(&Pl[wave][0], 32, 0, 0);
    o0 = wmma_bf16(pa, load_b_rm_bf16(Vp, 3 * DD, kb, 0),  o0);
    o1 = wmma_bf16(pa, load_b_rm_bf16(Vp, 3 * DD, kb, 16), o1);
    o2 = wmma_bf16(pa, load_b_rm_bf16(Vp, 3 * DD, kb, 32), o2);
    o3 = wmma_bf16(pa, load_b_rm_bf16(Vp, 3 * DD, kb, 48), o3);
    asm volatile("" ::: "memory");
  }
#pragma unroll
  for (int v = 0; v < 8; ++v) {
    int m = v + 8 * half;
    float inv = 1.f / ls[v];
    size_t rowo = ((size_t)b * TSEQ + q0 + m) * DD + h * HD;
    ob[rowo + 0 + n]  = (bf16)(o0[v] * inv);
    ob[rowo + 16 + n] = (bf16)(o1[v] * inv);
    ob[rowo + 32 + n] = (bf16)(o2[v] * inv);
    ob[rowo + 48 + n] = (bf16)(o3[v] * inv);
  }
}

// ---- LayerNorm: y = LN(x (+ f)) * s + b ; writes f32 and/or bf16 ------------
__global__ __launch_bounds__(256, 1)
void ln_kernel(const float* __restrict__ x, const float* __restrict__ f,
               const float* __restrict__ s, const float* __restrict__ bshift,
               float* __restrict__ outF, bf16* __restrict__ outB) {
  __shared__ float rs[8], rq[8];
  int row = blockIdx.x, tid = threadIdx.x;
  int wave = tid >> 5, lane = tid & 31;
  size_t base = (size_t)row * DD;
  float a0 = x[base + tid];
  float a1 = x[base + 256 + tid];
  if (f) { a0 += f[base + tid]; a1 += f[base + 256 + tid]; }
  float sum = a0 + a1, sq = a0 * a0 + a1 * a1;
#pragma unroll
  for (int m2 = 16; m2 >= 1; m2 >>= 1) {
    sum += __shfl_xor(sum, m2, 32);
    sq  += __shfl_xor(sq,  m2, 32);
  }
  if (lane == 0) { rs[wave] = sum; rq[wave] = sq; }
  __syncthreads();
  float S = 0.f, Q = 0.f;
#pragma unroll
  for (int w2 = 0; w2 < 8; ++w2) { S += rs[w2]; Q += rq[w2]; }
  float mean = S * (1.f / DD);
  float var  = Q * (1.f / DD) - mean * mean;
  float rstd = rsqrtf(var + 1e-5f);
  float y0 = (a0 - mean) * rstd * s[tid] + bshift[tid];
  float y1 = (a1 - mean) * rstd * s[256 + tid] + bshift[256 + tid];
  if (outF) { outF[base + tid] = y0; outF[base + 256 + tid] = y1; }
  if (outB) { outB[base + tid] = (bf16)y0; outB[base + 256 + tid] = (bf16)y1; }
}

// ---- hierarchical memory: WMMA scores + running top-4 + gather --------------
__global__ __launch_bounds__(128, 1)
void memread_kernel(const bf16* __restrict__ qmb,
                    const float* __restrict__ K0m, const float* __restrict__ V0m, const float* __restrict__ s0v,
                    const float* __restrict__ K1m, const float* __restrict__ V1m, const float* __restrict__ s1v,
                    const float* __restrict__ K2m, const float* __restrict__ V2m, const float* __restrict__ s2v,
                    float* __restrict__ readf) {
  __shared__ float Sl[4][256];
  __shared__ float Aw[4][16][4];
  __shared__ int   Iw[4][16][4];
  int wave = threadIdx.x >> 5, lane = threadIdx.x & 31;
  int n = lane & 15, half = lane >> 4;
  int q0 = (blockIdx.x * 4 + wave) * 16;

  bf16x16 qa[16];  // 16 query-row fragments covering K=512 (kept in VGPRs)
#pragma unroll
  for (int kk = 0; kk < 16; ++kk) qa[kk] = load_a_bf16(qmb, DD, q0, kk * 32);
  const float inv_sqrt_d = 0.04419417382f;  // 1/sqrt(512)

#pragma unroll 1
  for (int lvl = 0; lvl < 3; ++lvl) {
    const float* Km = lvl == 0 ? K0m : lvl == 1 ? K1m : K2m;
    const float* Vm = lvl == 0 ? V0m : lvl == 1 ? V1m : V2m;
    const float* sv = lvl == 0 ? s0v : lvl == 1 ? s1v : s2v;
    int S = lvl == 0 ? 16384 : lvl == 1 ? 8192 : 4096;

    float ts0 = -3e38f, ts1 = -3e38f, ts2 = -3e38f, ts3 = -3e38f;
    int ti0 = 0, ti1 = 0, ti2 = 0, ti3 = 0;

    for (int c0 = 0; c0 < S; c0 += 16) {
      floatx8 sc = {0, 0, 0, 0, 0, 0, 0, 0};
#pragma unroll
      for (int kk = 0; kk < 16; ++kk)
        sc = wmma_bf16(qa[kk], load_bT_f32(Km, DD, c0, kk * 32), sc);
      float saln = sv[c0 + n];
#pragma unroll
      for (int v = 0; v < 8; ++v)
        Sl[wave][(v + 8 * half) * 16 + n] = sc[v] * inv_sqrt_d + saln;
      asm volatile("s_wait_dscnt 0" ::: "memory");
      if (lane < 16) {  // lane r maintains top-4 for row r
#pragma unroll
        for (int j = 0; j < 16; ++j) {
          float val = Sl[wave][lane * 16 + j];
          int idx = c0 + j;
          if (val > ts3) {
            ts3 = val; ti3 = idx;
            if (ts3 > ts2) { float t = ts2; ts2 = ts3; ts3 = t; int i = ti2; ti2 = ti3; ti3 = i; }
            if (ts2 > ts1) { float t = ts1; ts1 = ts2; ts2 = t; int i = ti1; ti1 = ti2; ti2 = i; }
            if (ts1 > ts0) { float t = ts0; ts0 = ts1; ts1 = t; int i = ti0; ti0 = ti1; ti1 = i; }
          }
        }
      }
      asm volatile("" ::: "memory");
    }
    if (lane < 16) {  // softmax over 4 kept scores
      float e0 = __expf(ts0 - ts0), e1 = __expf(ts1 - ts0);
      float e2 = __expf(ts2 - ts0), e3 = __expf(ts3 - ts0);
      float r = 1.f / (e0 + e1 + e2 + e3);
      Aw[wave][lane][0] = e0 * r; Aw[wave][lane][1] = e1 * r;
      Aw[wave][lane][2] = e2 * r; Aw[wave][lane][3] = e3 * r;
      Iw[wave][lane][0] = ti0; Iw[wave][lane][1] = ti1;
      Iw[wave][lane][2] = ti2; Iw[wave][lane][3] = ti3;
    }
    asm volatile("s_wait_dscnt 0" ::: "memory");
    for (int r = 0; r < 16; ++r) {  // whole wave gathers V rows
      float a0 = Aw[wave][r][0], a1 = Aw[wave][r][1];
      float a2 = Aw[wave][r][2], a3 = Aw[wave][r][3];
      size_t i0 = (size_t)Iw[wave][r][0], i1 = (size_t)Iw[wave][r][1];
      size_t i2 = (size_t)Iw[wave][r][2], i3 = (size_t)Iw[wave][r][3];
      for (int d = lane; d < DD; d += 32) {
        float acc = a0 * Vm[i0 * DD + d] + a1 * Vm[i1 * DD + d] +
                    a2 * Vm[i2 * DD + d] + a3 * Vm[i3 * DD + d];
        size_t o = (size_t)(q0 + r) * DD + d;
        if (lvl == 0) readf[o] = acc; else readf[o] += acc;
      }
    }
  }
}

// ---- scale + f32 -> bf16 ----------------------------------------------------
__global__ void scale_bf16_kernel(const float* __restrict__ in, bf16* __restrict__ out,
                                  float s, int n) {
  int i = blockIdx.x * blockDim.x + threadIdx.x;
  if (i < n) out[i] = (bf16)(in[i] * s);
}

// ---- host launch ------------------------------------------------------------
extern "C" void kernel_launch(void* const* d_in, const int* in_sizes, int n_in,
                              void* d_out, int out_size, void* d_ws, size_t ws_size,
                              hipStream_t stream) {
  (void)in_sizes; (void)n_in; (void)out_size; (void)ws_size;
  const int*   ids  = (const int*)d_in[0];
  const float* tok  = (const float*)d_in[1];
  const float* pos  = (const float*)d_in[2];
  const float* Wqkv = (const float*)d_in[3];
  const float* bqkv = (const float*)d_in[4];
  const float* Wo   = (const float*)d_in[5];
  const float* bo   = (const float*)d_in[6];
  const float* ln1s = (const float*)d_in[7];
  const float* ln1b = (const float*)d_in[8];
  const float* W1   = (const float*)d_in[9];
  const float* b1   = (const float*)d_in[10];
  const float* W2   = (const float*)d_in[11];
  const float* b2   = (const float*)d_in[12];
  const float* ln2s = (const float*)d_in[13];
  const float* ln2b = (const float*)d_in[14];
  const float* Wq   = (const float*)d_in[15];
  const float* bq   = (const float*)d_in[16];
  const float* Wrp  = (const float*)d_in[17];
  const float* brp  = (const float*)d_in[18];
  const float* lnos = (const float*)d_in[19];
  const float* lnob = (const float*)d_in[20];
  const float* K0m  = (const float*)d_in[21];
  const float* V0m  = (const float*)d_in[22];
  const float* s0v  = (const float*)d_in[23];
  const float* K1m  = (const float*)d_in[24];
  const float* V1m  = (const float*)d_in[25];
  const float* s1v  = (const float*)d_in[26];
  const float* K2m  = (const float*)d_in[27];
  const float* V2m  = (const float*)d_in[28];
  const float* s2v  = (const float*)d_in[29];

  // workspace carving (~68 MB total)
  char* wsp = (char*)d_ws;
  size_t off = 0;
  auto alloc = [&](size_t bytes) {
    void* p = wsp + off;
    off += (bytes + 255) & ~(size_t)255;
    return p;
  };
  const size_t M = MTOT;
  float* xf    = (float*)alloc(M * DD * 4);
  bf16*  xb    = (bf16*) alloc(M * DD * 2);
  float* tmpf  = (float*)alloc(M * DD * 4);
  bf16*  qkvb  = (bf16*) alloc(M * 3 * DD * 2);
  bf16*  obuf  = (bf16*) alloc(M * DD * 2);
  bf16*  h1b   = (bf16*) alloc(M * 4 * DD * 2);
  bf16*  qmb   = (bf16*) alloc(M * DD * 2);
  float* readf = (float*)alloc(M * DD * 4);
  bf16*  readb = (bf16*) alloc(M * DD * 2);

  embed_kernel<<<M, 256, 0, stream>>>(ids, tok, pos, xf, xb);

  for (int i = 0; i < LL; ++i) {
    gemm_kernel<DD><<<dim3((3 * DD) / 128, M / 16), 256, 0, stream>>>(
        xb, Wqkv + (size_t)i * 3 * DD * DD, bqkv + (size_t)i * 3 * DD,
        nullptr, nullptr, qkvb, 3 * DD, 0);
    attn_kernel<<<dim3(TSEQ / 128, BB * HH), 256, 0, stream>>>(qkvb, obuf);
    gemm_kernel<DD><<<dim3(DD / 128, M / 16), 256, 0, stream>>>(
        obuf, Wo + (size_t)i * DD * DD, bo + (size_t)i * DD,
        nullptr, tmpf, nullptr, DD, 0);
    ln_kernel<<<M, 256, 0, stream>>>(xf, tmpf, ln1s + i * DD, ln1b + i * DD, xf, xb);
    gemm_kernel<DD><<<dim3((4 * DD) / 128, M / 16), 256, 0, stream>>>(
        xb, W1 + (size_t)i * 4 * DD * DD, b1 + (size_t)i * 4 * DD,
        nullptr, nullptr, h1b, 4 * DD, 1);
    gemm_kernel<4 * DD><<<dim3(DD / 128, M / 16), 256, 0, stream>>>(
        h1b, W2 + (size_t)i * DD * 4 * DD, b2 + (size_t)i * DD,
        nullptr, tmpf, nullptr, DD, 0);
    ln_kernel<<<M, 256, 0, stream>>>(xf, tmpf, ln2s + i * DD, ln2b + i * DD, xf, xb);
  }

  // memory query
  gemm_kernel<DD><<<dim3(DD / 128, M / 16), 256, 0, stream>>>(
      xb, Wq, bq, nullptr, nullptr, qmb, DD, 0);
  memread_kernel<<<M / 64, 128, 0, stream>>>(qmb, K0m, V0m, s0v, K1m, V1m, s1v,
                                             K2m, V2m, s2v, readf);
  scale_bf16_kernel<<<(int)(M * DD + 255) / 256, 256, 0, stream>>>(
      readf, readb, 1.f / 3.f, (int)(M * DD));
  // x = x + read @ Wrp^T + brp   (residual epilogue)
  gemm_kernel<DD><<<dim3(DD / 128, M / 16), 256, 0, stream>>>(
      readb, Wrp, brp, xf, xf, nullptr, DD, 0);
  // final LN -> bf16 only
  ln_kernel<<<M, 256, 0, stream>>>(xf, nullptr, lnos, lnob, nullptr, xb);
  // tied LM head: out = x @ tok_embed^T
  gemm_kernel<DD><<<dim3(VOC / 128, M / 16), 256, 0, stream>>>(
      xb, tok, nullptr, nullptr, (float*)d_out, nullptr, VOC, 0);
}
